// Attention_26989574488395
// MI455X (gfx1250) — compile-verified, matrix operand order checked by reading, not run on hardware
//
#include <hip/hip_runtime.h>

typedef unsigned short u16;
typedef __attribute__((ext_vector_type(16))) __bf16  v16bf;
typedef __attribute__((ext_vector_type(8)))  float   v8f;
typedef __attribute__((ext_vector_type(8)))  unsigned int v8u;
typedef __attribute__((ext_vector_type(4)))  int v4i;

#define AS1 __attribute__((address_space(1)))
#define AS3 __attribute__((address_space(3)))

#if defined(__has_builtin)
#if __has_builtin(__builtin_amdgcn_global_load_async_to_lds_b128)
#define HAVE_ASYNC_LDS 1
#endif
#endif

// ---------------- helpers ----------------

__device__ __forceinline__ u16 f2bf(float f) {
    unsigned u = __builtin_bit_cast(unsigned, f);
    unsigned r = u + 0x7FFFu + ((u >> 16) & 1u);   // round-to-nearest-even
    return (u16)(r >> 16);
}

__device__ __forceinline__ v8f v8f_zero() {
    v8f v;
#pragma unroll
    for (int i = 0; i < 8; ++i) v[i] = 0.0f;
    return v;
}

// 16B global -> LDS copy. Prefer gfx1250 GLOBAL_LOAD_ASYNC_TO_LDS_B128
// (ASYNCcnt-tracked, no VGPR round-trip); fall back to a regular copy.
__device__ __forceinline__ void cp_b128(const u16* g, u16* l) {
#if defined(HAVE_ASYNC_LDS)
    __builtin_amdgcn_global_load_async_to_lds_b128(
        (AS1 v4i*)g, (AS3 v4i*)l, 0, 0);
#else
    *reinterpret_cast<uint4*>(l) = *reinterpret_cast<const uint4*>(g);
#endif
}

__device__ __forceinline__ void cp_wait() {
#if defined(HAVE_ASYNC_LDS)
#if __has_builtin(__builtin_amdgcn_s_wait_asynccnt)
    __builtin_amdgcn_s_wait_asynccnt(0);
#else
    asm volatile("s_wait_asynccnt 0" ::: "memory");
#endif
#endif
}

// Load one 16x32 bf16 WMMA A/B fragment from a row-major buffer.
// rowPtr: pointer to this lane's row (row = base + (lane&15)).
// kBase:  chunkBase + (lane>>4)*8. ISA layout: elems 0..7 at kBase,
//         elems 8..15 at kBase+16 (16-bit A-matrix table, CDNA5 ISA).
__device__ __forceinline__ v16bf frag_ld(const u16* rowPtr, int kBase) {
    const uint4 lo = *reinterpret_cast<const uint4*>(rowPtr + kBase);
    const uint4 hi = *reinterpret_cast<const uint4*>(rowPtr + kBase + 16);
    v8u u;
    u[0] = lo.x; u[1] = lo.y; u[2] = lo.z; u[3] = lo.w;
    u[4] = hi.x; u[5] = hi.y; u[6] = hi.z; u[7] = hi.w;
    return __builtin_bit_cast(v16bf, u);
}

__device__ __forceinline__ v8f wmma_bf16(v16bf a, v16bf b, v8f c) {
    return __builtin_amdgcn_wmma_f32_16x16x32_bf16(
        false, a, false, b, (short)0, c, false, false);
}

// ---------------- kernel 1: fp32 -> bf16 convert ----------------

__global__ __launch_bounds__(256) void cvt_bf16_kernel(const float* __restrict__ src,
                                                       u16* __restrict__ dst) {
    int i = (blockIdx.x * 256 + threadIdx.x) * 4;
    float4 f = *reinterpret_cast<const float4*>(src + i);
    uint2 o;
    o.x = (unsigned)f2bf(f.x) | ((unsigned)f2bf(f.y) << 16);
    o.y = (unsigned)f2bf(f.z) | ((unsigned)f2bf(f.w) << 16);
    *reinterpret_cast<uint2*>(dst + i) = o;
}

// ---------------- kernel 2: fp32 [K][N] -> bf16 [N][K] transpose ----------------

__global__ __launch_bounds__(256) void transpose_cvt_kernel(const float* __restrict__ w,
                                                            u16* __restrict__ wT,
                                                            int K, int N) {
    __shared__ float tile[32][33];
    const int n0 = blockIdx.x * 32, k0 = blockIdx.y * 32;
    const int tx = threadIdx.x & 31, ty = threadIdx.x >> 5;   // 32 x 8
#pragma unroll
    for (int j = 0; j < 32; j += 8)
        tile[ty + j][tx] = w[(size_t)(k0 + ty + j) * N + n0 + tx];
    __syncthreads();
#pragma unroll
    for (int j = 0; j < 32; j += 8)
        wT[(size_t)(n0 + ty + j) * K + k0 + tx] = f2bf(tile[tx][ty + j]);
}

// ---------------- kernel 3: bf16 GEMM, C(f32)[M,N] = A[M,K] * BT[N,K]^T ----------------
// Compile-time N,K so column strides fold into immediate offsets.
// 128x128x32 block tile, 8 waves, each wave = 64x32 (4x2 WMMA tiles).

#define LDP 40   // 32 + 8 pad (keeps 16B row alignment, spreads LDS banks)

template <int N, int K>
__global__ __launch_bounds__(256) void gemm_bf16_kernel(const u16* __restrict__ A,
                                                        const u16* __restrict__ BT,
                                                        float* __restrict__ C) {
    __shared__ u16 As[128 * LDP];
    __shared__ u16 Bs[128 * LDP];
    const int tid  = threadIdx.x;
    const int lane = tid & 31, wid = tid >> 5;
    const int wm = wid >> 2, wn = wid & 3;         // 2 x 4 wave grid
    const int lr = lane & 15, lh = lane >> 4;
    const int tileM = blockIdx.y * 128, tileN = blockIdx.x * 128;

    v8f acc[4][2];
#pragma unroll
    for (int mi = 0; mi < 4; ++mi)
#pragma unroll
        for (int ni = 0; ni < 2; ++ni) acc[mi][ni] = v8f_zero();

    for (int k0 = 0; k0 < K; k0 += 32) {
#pragma unroll
        for (int i = 0; i < 2; ++i) {
            int idx = tid + i * 256;               // 512 x 16B chunks
            int row = idx >> 2, ch = idx & 3;      // 4 chunks of 8 elems per row
            cp_b128(&A [(size_t)(tileM + row) * K + k0 + ch * 8], &As[row * LDP + ch * 8]);
            cp_b128(&BT[(size_t)(tileN + row) * K + k0 + ch * 8], &Bs[row * LDP + ch * 8]);
        }
        cp_wait();
        __syncthreads();

        v16bf af[4], bfr[2];
#pragma unroll
        for (int mi = 0; mi < 4; ++mi)
            af[mi] = frag_ld(&As[(wm * 64 + mi * 16 + lr) * LDP], lh * 8);
#pragma unroll
        for (int ni = 0; ni < 2; ++ni)
            bfr[ni] = frag_ld(&Bs[(wn * 32 + ni * 16 + lr) * LDP], lh * 8);
#pragma unroll
        for (int mi = 0; mi < 4; ++mi)
#pragma unroll
            for (int ni = 0; ni < 2; ++ni)
                acc[mi][ni] = wmma_bf16(af[mi], bfr[ni], acc[mi][ni]);
        __syncthreads();
    }

    // epilogue: one base address per 16x16 tile, stores use immediate offsets
    const int row0 = tileM + wm * 64 + lh * 8;
    const int col0 = tileN + wn * 32 + lr;
#pragma unroll
    for (int mi = 0; mi < 4; ++mi)
#pragma unroll
        for (int ni = 0; ni < 2; ++ni) {
            float* Cp = C + (size_t)(row0 + mi * 16) * N + (col0 + ni * 16);
#pragma unroll
            for (int r = 0; r < 8; ++r)
                Cp[r * N] = acc[mi][ni][r];
        }
}

// ---------------- kernel 4: rotary + head split, qkv(f32) -> q/k/v bf16 [b,H,n,d] ----------------

__global__ __launch_bounds__(256) void rotary_split_kernel(const float* __restrict__ qkv,
                                                           const float* __restrict__ rope,
                                                           u16* __restrict__ qb,
                                                           u16* __restrict__ kb,
                                                           u16* __restrict__ vb) {
    const int gid = blockIdx.x * 256 + threadIdx.x;   // 2^21 threads
    const int i   = gid & 31;
    const int hd  = (gid >> 5) & 15;
    const int pos = (gid >> 9) & 2047;
    const int bi  = gid >> 20;

    const float* row = qkv + (size_t)(bi * 2048 + pos) * 3072;
    const int c = hd * 64 + i;
    float q1 = row[c],        q2 = row[c + 32];
    float k1 = row[1024 + c], k2 = row[1024 + c + 32];
    float v1 = row[2048 + c], v2 = row[2048 + c + 32];
    float r1 = rope[pos * 64 + i], r2 = rope[pos * 64 + i + 32];
    float c1 = __cosf(r1), s1 = __sinf(r1), c2 = __cosf(r2), s2 = __sinf(r2);

    size_t ob = ((size_t)(bi * 16 + hd) * 2048 + pos) * 64 + i;
    qb[ob]      = f2bf(q1 * c1 - q2 * s1);   // x*cos + rotate_half(x)*sin
    qb[ob + 32] = f2bf(q2 * c2 + q1 * s2);
    kb[ob]      = f2bf(k1 * c1 - k2 * s1);
    kb[ob + 32] = f2bf(k2 * c2 + k1 * s2);
    vb[ob]      = f2bf(v1);
    vb[ob + 32] = f2bf(v2);
}

// ---------------- kernel 5: flash attention ----------------
// grid = (n/128, b*H); 8 waves, each owns 16 query rows. Br=128, Bc=64, d=64.

#define KLD 72   // 64 + 8 pad

__global__ __launch_bounds__(256) void flash_attn_kernel(const u16* __restrict__ qb,
                                                         const u16* __restrict__ kb,
                                                         const u16* __restrict__ vb,
                                                         u16* __restrict__ ao) {
    __shared__ u16 Ks [64 * KLD];        // [key][d]
    __shared__ u16 VsT[64 * KLD];        // [d][key]
    __shared__ u16 Ps [8 * 16 * KLD];    // wave-private P re-layout scratch

    const int tid = threadIdx.x, lane = tid & 31, wid = tid >> 5;
    const int lr = lane & 15, lh = lane >> 4;
    const int bh = blockIdx.y;
    const int bi = bh >> 4, hd = bh & 15;
    const size_t headBase = (size_t)bh * 2048 * 64;
    const int q0 = blockIdx.x * 128 + wid * 16;

    // Q fragments stay in registers for the whole kernel
    const u16* qRow = qb + headBase + (size_t)(q0 + lr) * 64;
    const v16bf qf0 = frag_ld(qRow, 0  + lh * 8);
    const v16bf qf1 = frag_ld(qRow, 32 + lh * 8);

    v8f o[4];
    float mrow[8], lrow[8];
#pragma unroll
    for (int dt = 0; dt < 4; ++dt) o[dt] = v8f_zero();
#pragma unroll
    for (int r = 0; r < 8; ++r) { mrow[r] = -3.0e38f; lrow[r] = 0.0f; }

    const float kscale = 0.125f * 1.44269504f;   // 1/sqrt(64) * log2(e)

    for (int j0 = 0; j0 < 2048; j0 += 64) {
        // cooperative stage: K tile async to LDS (row-major), V tile transposed
#pragma unroll
        for (int i = 0; i < 2; ++i) {
            int idx = tid + i * 256;               // 512 x 8-elem chunks
            int row = idx >> 3, ch = idx & 7;
            cp_b128(&kb[headBase + (size_t)(j0 + row) * 64 + ch * 8],
                    &Ks[row * KLD + ch * 8]);
            uint4 v = *reinterpret_cast<const uint4*>(&vb[headBase + (size_t)(j0 + row) * 64 + ch * 8]);
            unsigned wv[4] = { v.x, v.y, v.z, v.w };
#pragma unroll
            for (int e = 0; e < 4; ++e) {
                VsT[(ch * 8 + 2 * e)     * KLD + row] = (u16)(wv[e] & 0xFFFFu);
                VsT[(ch * 8 + 2 * e + 1) * KLD + row] = (u16)(wv[e] >> 16);
            }
        }
        cp_wait();
        __syncthreads();

        // S = Q K^T (scaled into log2 domain)
        v8f st[4];
#pragma unroll
        for (int t = 0; t < 4; ++t) {
            v16bf kf0 = frag_ld(&Ks[(t * 16 + lr) * KLD], 0  + lh * 8);
            v16bf kf1 = frag_ld(&Ks[(t * 16 + lr) * KLD], 32 + lh * 8);
            v8f c = v8f_zero();
            c = wmma_bf16(qf0, kf0, c);
            c = wmma_bf16(qf1, kf1, c);
            st[t] = c * kscale;
        }

        // online softmax (rows spread over 16-lane halves; reduce across N lanes)
#pragma unroll
        for (int r = 0; r < 8; ++r) {
            float mx = fmaxf(fmaxf(st[0][r], st[1][r]), fmaxf(st[2][r], st[3][r]));
#pragma unroll
            for (int m = 1; m < 16; m <<= 1) mx = fmaxf(mx, __shfl_xor(mx, m, 32));
            float mnew  = fmaxf(mrow[r], mx);
            float alpha = exp2f(mrow[r] - mnew);
            mrow[r] = mnew;
            float rs = 0.0f;
#pragma unroll
            for (int t = 0; t < 4; ++t) {
                float p = exp2f(st[t][r] - mnew);
                st[t][r] = p;
                rs += p;
            }
#pragma unroll
            for (int m = 1; m < 16; m <<= 1) rs += __shfl_xor(rs, m, 32);
            lrow[r] = lrow[r] * alpha + rs;
#pragma unroll
            for (int dt = 0; dt < 4; ++dt) o[dt][r] *= alpha;
        }

        // re-layout P (C-frag -> A-frag) through wave-private LDS; DS ops are
        // in-order within a wave, so no barrier needed for this region.
        u16* pBase = &Ps[wid * 16 * KLD];
#pragma unroll
        for (int t = 0; t < 4; ++t)
#pragma unroll
            for (int r = 0; r < 8; ++r)
                pBase[(lh * 8 + r) * KLD + t * 16 + lr] = f2bf(st[t][r]);

        const u16* pRow = &pBase[lr * KLD];
        v16bf pf0 = frag_ld(pRow, 0  + lh * 8);
        v16bf pf1 = frag_ld(pRow, 32 + lh * 8);

        // O += P V
#pragma unroll
        for (int dt = 0; dt < 4; ++dt) {
            v16bf vf0 = frag_ld(&VsT[(dt * 16 + lr) * KLD], 0  + lh * 8);
            v16bf vf1 = frag_ld(&VsT[(dt * 16 + lr) * KLD], 32 + lh * 8);
            o[dt] = wmma_bf16(pf0, vf0, o[dt]);
            o[dt] = wmma_bf16(pf1, vf1, o[dt]);
        }
        __syncthreads();
    }

    // epilogue: normalize, write bf16 in [b, n, h] layout for out-projection
#pragma unroll
    for (int r = 0; r < 8; ++r) {
        float inv = 1.0f / lrow[r];
        int row = q0 + lh * 8 + r;
#pragma unroll
        for (int dt = 0; dt < 4; ++dt)
            ao[(size_t)(bi * 2048 + row) * 1024 + hd * 64 + dt * 16 + lr] =
                f2bf(o[dt][r] * inv);
    }
}

// ---------------- host side ----------------

extern "C" void kernel_launch(void* const* d_in, const int* in_sizes, int n_in,
                              void* d_out, int out_size, void* d_ws, size_t ws_size,
                              hipStream_t stream) {
    (void)in_sizes; (void)n_in; (void)out_size; (void)ws_size;
    const float* x    = (const float*)d_in[0];   // [2,2048,1024]
    const float* rope = (const float*)d_in[1];   // [2048,64]
    const float* wqkv = (const float*)d_in[2];   // [1024,3072]
    const float* wout = (const float*)d_in[3];   // [1024,1024]

    char* ws = (char*)d_ws;
    u16*   xb    = (u16*)(ws);                       // 8 MiB  [4096,1024] bf16
    u16*   wqkvT = (u16*)(ws + (8u   << 20));        // 6 MiB  [3072,1024] bf16
    u16*   woutT = (u16*)(ws + (14u  << 20));        // 2 MiB  [1024,1024] bf16
    float* qkvf  = (float*)(ws + (16u << 20));       // 48 MiB [4096,3072] f32
    u16*   qb    = (u16*)(ws + (64u  << 20));        // 8 MiB  [b,H,n,d] bf16
    u16*   kbuf  = (u16*)(ws + (72u  << 20));        // 8 MiB
    u16*   vbuf  = (u16*)(ws + (80u  << 20));        // 8 MiB
    u16*   ao    = (u16*)(ws + (88u  << 20));        // 8 MiB  [4096,1024] bf16

    // 1. x -> bf16
    cvt_bf16_kernel<<<(4096 * 1024) / (256 * 4), 256, 0, stream>>>(x, xb);
    // 2. weights -> bf16, transposed to [N][K]
    transpose_cvt_kernel<<<dim3(3072 / 32, 1024 / 32), 256, 0, stream>>>(wqkv, wqkvT, 1024, 3072);
    transpose_cvt_kernel<<<dim3(1024 / 32, 1024 / 32), 256, 0, stream>>>(wout, woutT, 1024, 1024);
    // 3. QKV projection (f32 accumulate)
    gemm_bf16_kernel<3072, 1024><<<dim3(3072 / 128, 4096 / 128), 256, 0, stream>>>(xb, wqkvT, qkvf);
    // 4. rotary + head split -> bf16 q/k/v
    rotary_split_kernel<<<(2 * 2048 * 16 * 32) / 256, 256, 0, stream>>>(qkvf, rope, qb, kbuf, vbuf);
    // 5. flash attention -> bf16 [b,n,h]
    flash_attn_kernel<<<dim3(2048 / 128, 2 * 16), 256, 0, stream>>>(qb, kbuf, vbuf, ao);
    // 6. output projection -> f32 d_out
    gemm_bf16_kernel<1024, 1024><<<dim3(1024 / 128, 4096 / 128), 256, 0, stream>>>(ao, woutT, (float*)d_out);
}
